// Qwen15MoeDecoderLayer_38774964748827
// MI455X (gfx1250) — matrix operand order repaired and checked by reading, not compiled, not run on hardware
//
#include <hip/hip_runtime.h>
#include <hip/hip_bf16.h>

// ---------------- constants ----------------
#define T_   1024
#define H_   2048
#define NH_  16
#define HD_  128
#define E_   16
#define TOPK_ 4
#define I_   1408
#define SI_  5632
#define EPS_ 1e-6f
#define SCALE_ 0.08838834764831845f   // 128^-0.5

// GEMM tiling
#define BM 128
#define BN 128
#define BK 32
#define LDSS 40   // bf16 elems per LDS row (32 + 8 pad), row = 80B (16B aligned)

#define USE_ASYNC_LDS 1

typedef __attribute__((ext_vector_type(16))) __bf16 v16bf;
typedef __attribute__((ext_vector_type(8)))  float  v8f;

union Frag16 { uint4 q[2]; v16bf v; };

__device__ __forceinline__ unsigned short f2bf(float f) {
    unsigned int u = __float_as_uint(f);
    u = u + 0x7fffu + ((u >> 16) & 1u);      // round-to-nearest-even
    return (unsigned short)(u >> 16);
}
__device__ __forceinline__ float bf2f(unsigned short h) {
    return __uint_as_float(((unsigned int)h) << 16);
}

// async global->LDS 16B copy (gfx1250 GLOBAL_LOAD_ASYNC_TO_LDS_B128, ASYNCcnt)
__device__ __forceinline__ void async_copy_b128(unsigned lds_off, const void* gaddr) {
#if USE_ASYNC_LDS
    asm volatile("global_load_async_to_lds_b128 %0, %1, off"
                 :: "v"(lds_off), "v"(gaddr) : "memory");
#else
    (void)lds_off; (void)gaddr;
#endif
}
__device__ __forceinline__ void async_wait_all() {
#if USE_ASYNC_LDS
    asm volatile("s_wait_asynccnt 0x0" ::: "memory");
#endif
}

// epilogue modes
#define EPI_STORE       0
#define EPI_BIAS        1
#define EPI_RESID       2
#define EPI_STORE_BF16  3
#define EPI_TSCALE      4
#define EPI_TSCALE_ACC  5

// ---------------------------------------------------------------------------
// Generic batched GEMM:  C[z] = A[z](bf16, MxK row-major) * B[z]
//   bmode 0: B is fp32 [K x N] row-major (converted to bf16 into LDS)
//   bmode 1: B is bf16 [N x K] row-major (i.e. B^T supplied directly)
// ---------------------------------------------------------------------------
__global__ __launch_bounds__(256) void gemm_bf16_kernel(
    const unsigned short* __restrict__ A, int lda, long long sA,
    const void* __restrict__ Bv, int ldb, long long sB, int bmode,
    void* __restrict__ Cv, int ldc, long long sC,
    int K,
    int epi,
    const float* __restrict__ bias,
    const float* __restrict__ resid,
    const float* __restrict__ tscale, int tss)
{
    __shared__ __align__(16) unsigned short As[BM * LDSS];
    __shared__ __align__(16) unsigned short Bs[BN * LDSS];

    const int tid  = threadIdx.x;
    const int lane = tid & 31;
    const int wave = tid >> 5;
    const int wm   = wave & 3;    // 4 waves along M
    const int wn   = wave >> 2;   // 2 waves along N
    const int z    = blockIdx.z;
    const int m0   = blockIdx.y * BM;
    const int n0   = blockIdx.x * BN;

    const unsigned short* Ab = A + (long long)z * sA;
    const float*          Bf = (const float*)Bv + (long long)z * sB;
    const unsigned short* Bt = (const unsigned short*)Bv + (long long)z * sB;

    v8f acc[2][4];
#pragma unroll
    for (int i = 0; i < 2; ++i)
#pragma unroll
        for (int j = 0; j < 4; ++j) acc[i][j] = (v8f)0.0f;

    // staging assignments
    const int arow = tid >> 1;           // A / bmode1: row, 16-elem column slab
    const int acol = (tid & 1) * 16;
    const int bn1  = tid & 127;          // bmode0: n column
    const int bkh  = (tid >> 7) * 16;    // bmode0: k slab start (0 or 16)
    const int lrow = lane & 15;
    const int lhalf = lane >> 4;

    const unsigned ldsA0 = (unsigned)(unsigned long long)&As[arow * LDSS + acol];
    const unsigned ldsB0 = (unsigned)(unsigned long long)&Bs[arow * LDSS + acol];

    for (int kb = 0; kb < K; kb += BK) {
        // ---- stage A tile (bf16 -> LDS) via async DMA ----
        {
            const unsigned short* src = Ab + (long long)(m0 + arow) * lda + kb + acol;
            async_copy_b128(ldsA0,      src);
            async_copy_b128(ldsA0 + 16, src + 8);
        }
        // ---- stage B tile as B^T (N-major) in LDS ----
        if (bmode == 0) {
            // coalesced column gather: thread owns one n, a 16-wide k slab
            const float* src = Bf + (long long)(kb + bkh) * ldb + n0 + bn1;
            union { unsigned short s[16]; uint4 q[2]; } u;
#pragma unroll
            for (int kk = 0; kk < 16; ++kk)
                u.s[kk] = f2bf(src[(long long)kk * ldb]);
            *(uint4*)&Bs[bn1 * LDSS + bkh]     = u.q[0];
            *(uint4*)&Bs[bn1 * LDSS + bkh + 8] = u.q[1];
            if (kb + BK < K)
                __builtin_prefetch(src + (long long)BK * ldb, 0, 0);
        } else {
            const unsigned short* src = Bt + (long long)(n0 + arow) * ldb + kb + acol;
            async_copy_b128(ldsB0,      src);
            async_copy_b128(ldsB0 + 16, src + 8);
        }
        async_wait_all();
        __syncthreads();

        // ---- fragments (ISA 16-bit A 16x32 / B 32x16 wave32 layouts) ----
        Frag16 af[2], bfr[4];
#pragma unroll
        for (int i = 0; i < 2; ++i) {
            int row = wm * 32 + i * 16 + lrow;
            int cb  = lhalf * 8;                 // lanes16-31: K 8..15 / 24..31
            af[i].q[0] = *(const uint4*)&As[row * LDSS + cb];
            af[i].q[1] = *(const uint4*)&As[row * LDSS + cb + 16];
        }
#pragma unroll
        for (int j = 0; j < 4; ++j) {
            int col = wn * 64 + j * 16 + lrow;
            int ks  = lhalf * 8;                 // lanes16-31: K 8..15 / 24..31
            bfr[j].q[0] = *(const uint4*)&Bs[col * LDSS + ks];
            bfr[j].q[1] = *(const uint4*)&Bs[col * LDSS + ks + 16];
        }
#pragma unroll
        for (int i = 0; i < 2; ++i)
#pragma unroll
            for (int j = 0; j < 4; ++j)
                acc[i][j] = __builtin_amdgcn_wmma_f32_16x16x32_bf16(
                    false, af[i].v, false, bfr[j].v,
                    (short)0, acc[i][j], false, false);
        __syncthreads();
    }

    // ---- epilogue ----
    float*          Cf = (float*)Cv + (long long)z * sC;
    unsigned short* Cb = (unsigned short*)Cv + (long long)z * sC;
#pragma unroll
    for (int i = 0; i < 2; ++i)
#pragma unroll
        for (int j = 0; j < 4; ++j)
#pragma unroll
            for (int r = 0; r < 8; ++r) {
                int row = m0 + wm * 32 + i * 16 + lhalf * 8 + r;
                int col = n0 + wn * 64 + j * 16 + lrow;
                float v = acc[i][j][r];
                long long idx = (long long)row * ldc + col;
                switch (epi) {
                    case EPI_STORE:      Cf[idx] = v; break;
                    case EPI_BIAS:       Cf[idx] = v + bias[col]; break;
                    case EPI_RESID:      Cf[idx] = v + resid[idx]; break;
                    case EPI_STORE_BF16: Cb[idx] = f2bf(v); break;
                    case EPI_TSCALE:     Cf[idx] = v * tscale[(long long)row * tss]; break;
                    case EPI_TSCALE_ACC: Cf[idx] += v * tscale[(long long)row * tss]; break;
                }
            }
}

// ---------------------------------------------------------------------------
// RMSNorm: fp32 in -> bf16 out, optional residual copy to fp32 output
// ---------------------------------------------------------------------------
__global__ __launch_bounds__(256) void rmsnorm_kernel(
    const float* __restrict__ x, const float* __restrict__ w,
    unsigned short* __restrict__ out_bf, float* __restrict__ resid_copy)
{
    int t = blockIdx.x;
    const float* row = x + (long long)t * H_;
    __shared__ float red[256];
    float s = 0.0f;
    for (int i = threadIdx.x; i < H_; i += 256) { float v = row[i]; s += v * v; }
    red[threadIdx.x] = s; __syncthreads();
    for (int off = 128; off > 0; off >>= 1) {
        if (threadIdx.x < off) red[threadIdx.x] += red[threadIdx.x + off];
        __syncthreads();
    }
    float scale = rsqrtf(red[0] / (float)H_ + EPS_);
    for (int i = threadIdx.x; i < H_; i += 256) {
        float v = row[i];
        out_bf[(long long)t * H_ + i] = f2bf(v * scale * w[i]);
        if (resid_copy) resid_copy[(long long)t * H_ + i] = v;
    }
}

// ---------------------------------------------------------------------------
// RoPE: read q,k from fp32 qkv buffer, write bf16 [NH][T][HD]
// ---------------------------------------------------------------------------
__global__ __launch_bounds__(64) void rope_kernel(
    const int* __restrict__ pos, const float* __restrict__ qkv,
    unsigned short* __restrict__ qb, unsigned short* __restrict__ kb)
{
    int t = blockIdx.x, h = blockIdx.y, d = threadIdx.x;   // d in 0..63
    float p   = (float)pos[t];
    float inv = __powf(1.0e6f, -((float)(2 * d) / (float)HD_));
    float ang = p * inv;
    float c = __cosf(ang), s = __sinf(ang);
    const float* q = qkv + (long long)t * (3 * H_) + h * HD_;
    const float* k = q + H_;
    float q1 = q[d], q2 = q[d + 64], k1 = k[d], k2 = k[d + 64];
    long long o = ((long long)h * T_ + t) * HD_;
    qb[o + d]      = f2bf(q1 * c - q2 * s);
    qb[o + d + 64] = f2bf(q2 * c + q1 * s);
    kb[o + d]      = f2bf(k1 * c - k2 * s);
    kb[o + d + 64] = f2bf(k2 * c + k1 * s);
}

// ---------------------------------------------------------------------------
// Causal masked softmax over scores rows; writes bf16 probabilities
// ---------------------------------------------------------------------------
__global__ __launch_bounds__(256) void softmax_kernel(
    const float* __restrict__ scores, unsigned short* __restrict__ p,
    const int* __restrict__ pos)
{
    int q = blockIdx.x, h = blockIdx.y;
    const float* row = scores + ((long long)h * T_ + q) * T_;
    unsigned short* prow = p + ((long long)h * T_ + q) * T_;
    int pq = pos[q];
    __shared__ float red[256];
    float vals[4];
    float mx = -3.0e38f;
#pragma unroll
    for (int j = 0; j < 4; ++j) {
        int k = j * 256 + threadIdx.x;
        float v = row[k] * SCALE_;
        if (pos[k] > pq) v = -3.0e38f;
        vals[j] = v;
        mx = fmaxf(mx, v);
    }
    red[threadIdx.x] = mx; __syncthreads();
    for (int off = 128; off > 0; off >>= 1) {
        if (threadIdx.x < off) red[threadIdx.x] = fmaxf(red[threadIdx.x], red[threadIdx.x + off]);
        __syncthreads();
    }
    float m = red[0]; __syncthreads();
    float se = 0.0f;
#pragma unroll
    for (int j = 0; j < 4; ++j) { float e = __expf(vals[j] - m); vals[j] = e; se += e; }
    red[threadIdx.x] = se; __syncthreads();
    for (int off = 128; off > 0; off >>= 1) {
        if (threadIdx.x < off) red[threadIdx.x] += red[threadIdx.x + off];
        __syncthreads();
    }
    float invs = 1.0f / red[0];
#pragma unroll
    for (int j = 0; j < 4; ++j) {
        int k = j * 256 + threadIdx.x;
        prow[k] = f2bf(vals[j] * invs);
    }
}

// ---------------------------------------------------------------------------
// Router: logits, softmax, top-4 renorm -> comb[T][E]; shared-gate sigmoid
// ---------------------------------------------------------------------------
__global__ __launch_bounds__(256) void gate_kernel(
    const unsigned short* __restrict__ x2, const float* __restrict__ gate_w,
    const float* __restrict__ sh_gate, float* __restrict__ comb,
    float* __restrict__ shsig)
{
    int t = blockIdx.x;
    const unsigned short* row = x2 + (long long)t * H_;
    float part[17];
#pragma unroll
    for (int e = 0; e < 17; ++e) part[e] = 0.0f;
    for (int i = threadIdx.x; i < H_; i += 256) {
        float xv = bf2f(row[i]);
#pragma unroll
        for (int e = 0; e < 16; ++e) part[e] += xv * gate_w[i * E_ + e];
        part[16] += xv * sh_gate[i];
    }
    __shared__ float red[256];
    __shared__ float res[17];
    for (int e = 0; e < 17; ++e) {
        red[threadIdx.x] = part[e]; __syncthreads();
        for (int off = 128; off > 0; off >>= 1) {
            if (threadIdx.x < off) red[threadIdx.x] += red[threadIdx.x + off];
            __syncthreads();
        }
        if (threadIdx.x == 0) res[e] = red[0];
        __syncthreads();
    }
    if (threadIdx.x == 0) {
        float mx = res[0];
        for (int e = 1; e < 16; ++e) mx = fmaxf(mx, res[e]);
        float pr[16]; float se = 0.0f;
        for (int e = 0; e < 16; ++e) { pr[e] = __expf(res[e] - mx); se += pr[e]; }
        for (int e = 0; e < 16; ++e) pr[e] /= se;
        int   id4[TOPK_]; float w4[TOPK_];
        bool used[16];
        for (int e = 0; e < 16; ++e) used[e] = false;
        for (int s = 0; s < TOPK_; ++s) {
            int bi = -1; float bv = -1.0f;
            for (int e = 0; e < 16; ++e)
                if (!used[e] && pr[e] > bv) { bv = pr[e]; bi = e; }
            used[bi] = true; id4[s] = bi; w4[s] = bv;
        }
        float sw = w4[0] + w4[1] + w4[2] + w4[3];
        for (int e = 0; e < 16; ++e) comb[(long long)t * E_ + e] = 0.0f;
        for (int s = 0; s < TOPK_; ++s) comb[(long long)t * E_ + id4[s]] = w4[s] / sw;
        shsig[t] = 1.0f / (1.0f + __expf(-res[16]));
    }
}

// ---------------------------------------------------------------------------
// SiLU(g)*u : fp32 [T][2*Iw] -> bf16 [T][Iw]
// ---------------------------------------------------------------------------
__global__ __launch_bounds__(256) void silu_mul_kernel(
    const float* __restrict__ gu, unsigned short* __restrict__ act,
    int Iw, long long total)
{
    long long idx = (long long)blockIdx.x * 256 + threadIdx.x;
    if (idx >= total) return;
    long long t = idx / Iw;
    int i = (int)(idx - t * Iw);
    float g = gu[t * (2LL * Iw) + i];
    float u = gu[t * (2LL * Iw) + Iw + i];
    act[idx] = f2bf((g / (1.0f + __expf(-g))) * u);
}

// ---------------------------------------------------------------------------
extern "C" void kernel_launch(void* const* d_in, const int* in_sizes, int n_in,
                              void* d_out, int out_size, void* d_ws, size_t ws_size,
                              hipStream_t stream) {
    (void)in_sizes; (void)n_in; (void)out_size; (void)ws_size;
    const int*   positions = (const int*)d_in[0];
    const float* hidden    = (const float*)d_in[1];
    const float* ln1_w     = (const float*)d_in[2];
    const float* ln2_w     = (const float*)d_in[3];
    const float* qkv_w     = (const float*)d_in[4];
    const float* qkv_b     = (const float*)d_in[5];
    const float* o_w       = (const float*)d_in[6];
    const float* gate_w    = (const float*)d_in[7];
    const float* e_gu_w    = (const float*)d_in[8];
    const float* e_dn_w    = (const float*)d_in[9];
    const float* sh_gu_w   = (const float*)d_in[10];
    const float* sh_dn_w   = (const float*)d_in[11];
    const float* sh_gate_w = (const float*)d_in[12];

    // ---- workspace layout (bytes, 256-aligned) ----
    char* ws = (char*)d_ws;
    size_t off = 0;
    auto alloc = [&](size_t bytes) { size_t r = off; off = (off + bytes + 255) & ~(size_t)255; return r; };
    size_t oXB   = alloc((size_t)T_ * H_ * 2);            // x1 bf16
    size_t oQKV  = alloc((size_t)T_ * 3 * H_ * 4);        // qkv fp32
    size_t oQB   = alloc((size_t)NH_ * T_ * HD_ * 2);     // q roped bf16 [h][t][d]
    size_t oKB   = alloc((size_t)NH_ * T_ * HD_ * 2);     // k roped bf16 [h][t][d]
    size_t oBIG0 = alloc((size_t)NH_ * T_ * T_ * 4);      // scores f32; later shared gu f32
    size_t oBIG1 = alloc((size_t)NH_ * T_ * T_ * 2);      // P bf16; later MoE scratch
    size_t oAO   = alloc((size_t)T_ * H_ * 2);            // attn out bf16
    size_t oRES2 = alloc((size_t)T_ * H_ * 4);            // residual2 f32
    size_t oX2   = alloc((size_t)T_ * H_ * 2);            // x2 bf16
    size_t oCOMB = alloc((size_t)T_ * E_ * 4);
    size_t oSSIG = alloc((size_t)T_ * 4);

    unsigned short* X1   = (unsigned short*)(ws + oXB);
    float*          QKV  = (float*)(ws + oQKV);
    unsigned short* QB   = (unsigned short*)(ws + oQB);
    unsigned short* KB   = (unsigned short*)(ws + oKB);
    float*          SCOR = (float*)(ws + oBIG0);
    unsigned short* PRB  = (unsigned short*)(ws + oBIG1);
    unsigned short* AO   = (unsigned short*)(ws + oAO);
    float*          RES2 = (float*)(ws + oRES2);
    unsigned short* X2   = (unsigned short*)(ws + oX2);
    float*          COMB = (float*)(ws + oCOMB);
    float*          SSIG = (float*)(ws + oSSIG);
    // MoE scratch carved from BIG0/BIG1 (reused after attention)
    float*          SHGU = (float*)(ws + oBIG0);                       // T*2SI f32 = 46MB (<64MB)
    unsigned short* SHAC = (unsigned short*)(ws + oBIG1);              // T*SI bf16 = 11.5MB
    float*          EGU  = (float*)(ws + oBIG1 + (12u << 20));         // T*2I f32 = 11.5MB
    unsigned short* EAC  = (unsigned short*)(ws + oBIG1 + (24u << 20));// T*I bf16 = 2.9MB

    float* out_moe   = (float*)d_out;             // [T,H]
    float* out_resid = (float*)d_out + (size_t)T_ * H_;

    // 1) rmsnorm1 -> x1 bf16
    rmsnorm_kernel<<<T_, 256, 0, stream>>>(hidden, ln1_w, X1, nullptr);

    // 2) QKV GEMM: [T,H] x [H,3H] + b -> QKV f32
    gemm_bf16_kernel<<<dim3(3 * H_ / BN, T_ / BM, 1), 256, 0, stream>>>(
        X1, H_, 0, qkv_w, 3 * H_, 0, 0, QKV, 3 * H_, 0, H_,
        EPI_BIAS, qkv_b, nullptr, nullptr, 0);

    // 3) RoPE -> QB/KB bf16 per-head
    rope_kernel<<<dim3(T_, NH_), 64, 0, stream>>>(positions, QKV, QB, KB);

    // 4) scores = Q K^T (batched over heads; K supplied as B^T bf16)
    gemm_bf16_kernel<<<dim3(T_ / BN, T_ / BM, NH_), 256, 0, stream>>>(
        QB, HD_, (long long)T_ * HD_,
        KB, HD_, (long long)T_ * HD_, 1,
        SCOR, T_, (long long)T_ * T_, HD_,
        EPI_STORE, nullptr, nullptr, nullptr, 0);

    // 5) masked softmax -> P bf16
    softmax_kernel<<<dim3(T_, NH_), 256, 0, stream>>>(SCOR, PRB, positions);

    // 6) attn_out = P V (V read in place from fp32 QKV buffer), bf16 out [T,H]
    gemm_bf16_kernel<<<dim3(HD_ / BN, T_ / BM, NH_), 256, 0, stream>>>(
        PRB, T_, (long long)T_ * T_,
        QKV + 2 * H_, 3 * H_, (long long)HD_, 0,
        AO, H_, (long long)HD_, T_,
        EPI_STORE_BF16, nullptr, nullptr, nullptr, 0);

    // 7) O proj + residual -> RES2 f32
    gemm_bf16_kernel<<<dim3(H_ / BN, T_ / BM, 1), 256, 0, stream>>>(
        AO, H_, 0, o_w, H_, 0, 0, RES2, H_, 0, H_,
        EPI_RESID, nullptr, hidden, nullptr, 0);

    // 8) rmsnorm2 -> X2 bf16, and copy residual to output slot 2
    rmsnorm_kernel<<<T_, 256, 0, stream>>>(RES2, ln2_w, X2, out_resid);

    // 9) router + shared-gate sigmoid
    gate_kernel<<<T_, 256, 0, stream>>>(X2, gate_w, sh_gate_w, COMB, SSIG);

    // 10) shared expert gate_up -> SHGU f32
    gemm_bf16_kernel<<<dim3(2 * SI_ / BN, T_ / BM, 1), 256, 0, stream>>>(
        X2, H_, 0, sh_gu_w, 2 * SI_, 0, 0, SHGU, 2 * SI_, 0, H_,
        EPI_STORE, nullptr, nullptr, nullptr, 0);

    // 11) silu*mul -> SHAC bf16
    {
        long long total = (long long)T_ * SI_;
        silu_mul_kernel<<<(unsigned)((total + 255) / 256), 256, 0, stream>>>(SHGU, SHAC, SI_, total);
    }

    // 12) shared down, scaled by sigmoid gate -> out_moe (initializes output)
    gemm_bf16_kernel<<<dim3(H_ / BN, T_ / BM, 1), 256, 0, stream>>>(
        SHAC, SI_, 0, sh_dn_w, H_, 0, 0, out_moe, H_, 0, SI_,
        EPI_TSCALE, nullptr, nullptr, SSIG, 1);

    // 13) routed experts: gate_up -> silu -> down, accumulate with routing weight
    for (int e = 0; e < E_; ++e) {
        gemm_bf16_kernel<<<dim3(2 * I_ / BN, T_ / BM, 1), 256, 0, stream>>>(
            X2, H_, 0, e_gu_w + (size_t)e * H_ * 2 * I_, 2 * I_, 0, 0,
            EGU, 2 * I_, 0, H_,
            EPI_STORE, nullptr, nullptr, nullptr, 0);
        long long total = (long long)T_ * I_;
        silu_mul_kernel<<<(unsigned)((total + 255) / 256), 256, 0, stream>>>(EGU, EAC, I_, total);
        gemm_bf16_kernel<<<dim3(H_ / BN, T_ / BM, 1), 256, 0, stream>>>(
            EAC, I_, 0, e_dn_w + (size_t)e * I_ * H_, H_, 0, 0,
            out_moe, H_, 0, I_,
            EPI_TSCALE_ACC, nullptr, nullptr, COMB + e, E_);
    }
}
